// Affinity_13082470384087
// MI455X (gfx1250) — matrix operand order repaired and checked by reading, not compiled, not run on hardware
//
#include <hip/hip_runtime.h>
#include <stdint.h>

#define NN 8192
#define DD 512

typedef __attribute__((ext_vector_type(16))) __bf16 v16bf;
typedef __attribute__((ext_vector_type(8)))  __bf16 v8bf;
typedef __attribute__((ext_vector_type(8)))  float  v8f;

// ---------- bf16 helpers (RNE, no NaN inputs expected) ----------
static __device__ __forceinline__ __bf16 f2bf(float f) {
  unsigned int u = __float_as_uint(f);
  u += 0x7FFFu + ((u >> 16) & 1u);
  unsigned short hv = (unsigned short)(u >> 16);
  __bf16 r;
  __builtin_memcpy(&r, &hv, 2);
  return r;
}
static __device__ __forceinline__ float bf2f(__bf16 h) {
  unsigned short hv;
  __builtin_memcpy(&hv, &h, 2);
  return __uint_as_float(((unsigned int)hv) << 16);
}

// ---------- Kernel 1: row norms + bf16 hi/lo split ----------
__global__ void prep_kernel(const float* __restrict__ x,
                            float* __restrict__ x2,
                            __bf16* __restrict__ ahi,
                            __bf16* __restrict__ alo)
{
  __shared__ float red[256];
  const int row = blockIdx.x;
  const int tid = threadIdx.x;
  float s = 0.f;
  for (int k = tid; k < DD; k += 256) {
    float v = x[(size_t)row * DD + k];
    __bf16 h = f2bf(v);
    __bf16 l = f2bf(v - bf2f(h));
    ahi[(size_t)row * DD + k] = h;
    alo[(size_t)row * DD + k] = l;
    s += v * v;
  }
  red[tid] = s;
  __syncthreads();
  for (int st = 128; st > 0; st >>= 1) {
    if (tid < st) red[tid] += red[tid + st];
    __syncthreads();
  }
  if (tid == 0) x2[row] = red[0];
}

// ---------- shared 16x64 WMMA strip: acc = X[rows] . X[cols]^T ----------
static __device__ __forceinline__ void gemm_strip(const __bf16* __restrict__ ahi,
                                                  const __bf16* __restrict__ alo,
                                                  int rowbase, int colbase, int lane,
                                                  v8f acc[4])
{
  const int m = lane & 15;
  const int h = lane >> 4;
  const __bf16* Arh = ahi + (size_t)(rowbase + m) * DD;
  const __bf16* Arl = alo + (size_t)(rowbase + m) * DD;
  for (int kc = 0; kc < DD / 32; ++kc) {
    const int kb = kc * 32;
    // A fragment (16x32 bf16): elems 0-15 = K[kb+16h .. +16)
    v8bf p = *(const v8bf*)(Arh + kb + h * 16);
    v8bf q = *(const v8bf*)(Arh + kb + h * 16 + 8);
    v16bf a_h = __builtin_shufflevector(p, q, 0,1,2,3,4,5,6,7,8,9,10,11,12,13,14,15);
    p = *(const v8bf*)(Arl + kb + h * 16);
    q = *(const v8bf*)(Arl + kb + h * 16 + 8);
    v16bf a_l = __builtin_shufflevector(p, q, 0,1,2,3,4,5,6,7,8,9,10,11,12,13,14,15);
#pragma unroll
    for (int t = 0; t < 4; ++t) {
      // B fragment (32x16 bf16): column n=m is row (colbase+t*16+m) of X; elems = K[kb+16h .. +16)
      const __bf16* Bh = ahi + (size_t)(colbase + t * 16 + m) * DD + kb + h * 16;
      const __bf16* Bl = alo + (size_t)(colbase + t * 16 + m) * DD + kb + h * 16;
      v16bf b_h = *(const v16bf*)Bh;
      v16bf b_l = *(const v16bf*)Bl;
      acc[t] = __builtin_amdgcn_wmma_f32_16x16x32_bf16(false, a_h, false, b_h, (short)0, acc[t], false, false);
      acc[t] = __builtin_amdgcn_wmma_f32_16x16x32_bf16(false, a_h, false, b_l, (short)0, acc[t], false, false);
      acc[t] = __builtin_amdgcn_wmma_f32_16x16x32_bf16(false, a_l, false, b_h, (short)0, acc[t], false, false);
    }
  }
}

// ---------- Kernel 2: fused dist + per-row top-10 ----------
__global__ void topk_kernel(const __bf16* __restrict__ ahi,
                            const __bf16* __restrict__ alo,
                            const float* __restrict__ x2,
                            float* __restrict__ topd,
                            int* __restrict__ topi)
{
  __shared__ float lds_dist[4][16][64];   // per-wave staging of a 16x64 dist strip
  __shared__ float mgd[8][16][10];        // 8 partial lists per row (4 waves x 2 half-waves)
  __shared__ int   mgi[8][16][10];

  const int tid  = threadIdx.x;
  const int wave = tid >> 5;
  const int lane = tid & 31;
  const int m = lane & 15;
  const int h = lane >> 4;
  const int rowbase = blockIdx.x * 16;

  float best[10];
  int   bidx[10];
#pragma unroll
  for (int t = 0; t < 10; ++t) { best[t] = 3.0e38f; bidx[t] = 0; }

  float x2r[8];
#pragma unroll
  for (int v = 0; v < 8; ++v) x2r[v] = x2[rowbase + v + 8 * h];

  for (int strip = wave; strip < NN / 64; strip += 4) {
    const int colbase = strip * 64;
    v8f acc[4] = {};
    gemm_strip(ahi, alo, rowbase, colbase, lane, acc);

    // dist = x2_i + x2_j - 2*G  -> LDS (intra-wave LDS ops are in-order)
#pragma unroll
    for (int t = 0; t < 4; ++t) {
      const float x2c = x2[colbase + t * 16 + m];
#pragma unroll
      for (int v = 0; v < 8; ++v) {
        const float dd = x2r[v] + x2c - 2.0f * acc[t][v];
        lds_dist[wave][v + 8 * h][t * 16 + m] = dd;
      }
    }

    // all 32 lanes: lane handles row m, column half h (32 candidates each)
    const int rowg = rowbase + m;
    for (int i = 0; i < 32; ++i) {
      const int cc = h * 32 + i;
      const int colg = colbase + cc;
      const float dd = lds_dist[wave][m][cc];
      if (colg == rowg) continue;
      if (dd < best[9]) {                  // sorted insert (register resident)
        best[9] = dd; bidx[9] = colg;
#pragma unroll
        for (int t = 9; t >= 1; --t) {
          if (best[t] < best[t - 1]) {
            float tf = best[t]; best[t] = best[t - 1]; best[t - 1] = tf;
            int   ti = bidx[t]; bidx[t] = bidx[t - 1]; bidx[t - 1] = ti;
          }
        }
      }
    }
  }

  const int slot = wave * 2 + h;          // 0..7 partial lists per row
#pragma unroll
  for (int t = 0; t < 10; ++t) { mgd[slot][m][t] = best[t]; mgi[slot][m][t] = bidx[t]; }
  __syncthreads();

  // wave 0, lanes 0-15: merge 8 sorted lists of 10 -> global top-10 per row
  if (wave == 0 && lane < 16) {
    int hp[8] = {0,0,0,0,0,0,0,0};
    const int rowg = rowbase + lane;
    for (int o = 0; o < 10; ++o) {
      float mn = 3.4e38f; int mw = 0;
#pragma unroll
      for (int w = 0; w < 8; ++w) {
        const float v = mgd[w][lane][hp[w]];
        if (v < mn) { mn = v; mw = w; }
      }
      topd[rowg * 10 + o] = mn;
      topi[rowg * 10 + o] = mgi[mw][lane][hp[mw]];
      hp[mw]++;
    }
  }
}

// ---------- Kernel 3: zero the column mask ----------
__global__ void zmask_kernel(float* __restrict__ mask)
{
  mask[blockIdx.x * 256 + threadIdx.x] = 0.0f;
}

// ---------- Kernel 4: median of 3rd-NN dists -> c = 1/(2*sigma^2) ----------
__global__ void median_kernel(const float* __restrict__ topd, float* __restrict__ csc)
{
  __shared__ float s[NN];                 // 32 KB of the 320 KB LDS
  const int tid = threadIdx.x;
  for (int i = tid; i < NN; i += 1024) s[i] = topd[i * 10 + 2];
  __syncthreads();
  for (int k = 2; k <= NN; k <<= 1) {
    for (int j = k >> 1; j > 0; j >>= 1) {
      for (int i = tid; i < NN; i += 1024) {
        const int ixj = i ^ j;
        if (ixj > i) {
          const float a = s[i], b = s[ixj];
          const bool up = ((i & k) == 0);
          if (up ? (a > b) : (a < b)) { s[i] = b; s[ixj] = a; }
        }
      }
      __syncthreads();
    }
  }
  if (tid == 0) {
    const float med = 0.5f * (s[NN / 2 - 1] + s[NN / 2]);  // jnp.median, even count
    csc[0] = 1.0f / (2.0f * med);                          // sigma^2 == med
  }
}

// ---------- Kernel 5: scatter mask[idx] = 1 (benign races) ----------
__global__ void scatter_kernel(const int* __restrict__ topi, float* __restrict__ mask)
{
  const int g = blockIdx.x * 256 + threadIdx.x;            // g < N*10
  mask[topi[g]] = 1.0f;
}

// ---------- Kernel 6: fused dist + exp + mask + symmetrize ----------
__global__ void affinity_kernel(const __bf16* __restrict__ ahi,
                                const __bf16* __restrict__ alo,
                                const float* __restrict__ x2,
                                const float* __restrict__ mask,
                                const float* __restrict__ csc,
                                float* __restrict__ out)
{
  const int tid  = threadIdx.x;
  const int wave = tid >> 5;
  const int lane = tid & 31;
  const int m = lane & 15;
  const int h = lane >> 4;
  const int rowbase = blockIdx.y * 64 + wave * 16;
  const int colbase = blockIdx.x * 64;

  v8f acc[4] = {};
  gemm_strip(ahi, alo, rowbase, colbase, lane, acc);

  const float c = csc[0];
  float x2r[8], mr[8];
#pragma unroll
  for (int v = 0; v < 8; ++v) {
    const int rg = rowbase + v + 8 * h;
    x2r[v] = x2[rg];
    mr[v]  = mask[rg];
  }
#pragma unroll
  for (int t = 0; t < 4; ++t) {
    const int cg = colbase + t * 16 + m;
    const float x2c = x2[cg];
    const float mc  = mask[cg];
#pragma unroll
    for (int v = 0; v < 8; ++v) {
      const int rg = rowbase + v + 8 * h;
      const float dd = x2r[v] + x2c - 2.0f * acc[t][v];
      float val = 0.5f * __expf(-dd * c) * (mr[v] + mc);   // sym = 0.5*(aff+aff^T)
      if (rg == cg) val = 0.0f;                            // diagonal: dist=1e9 -> 0
      out[(size_t)rg * NN + cg] = val;
    }
  }
}

// ---------- host launcher ----------
extern "C" void kernel_launch(void* const* d_in, const int* in_sizes, int n_in,
                              void* d_out, int out_size, void* d_ws, size_t ws_size,
                              hipStream_t stream)
{
  (void)in_sizes; (void)n_in; (void)out_size; (void)ws_size;
  const float* x = (const float*)d_in[0];
  float* out = (float*)d_out;

  char* ws = (char*)d_ws;
  // workspace layout (~17.5 MB total, all offsets 32B-aligned)
  float*  x2   = (float*) (ws);                                    // 32 KB
  __bf16* ahi  = (__bf16*)(ws + 32768);                            // 8 MB
  __bf16* alo  = (__bf16*)(ws + 32768 + 8388608);                  // 8 MB
  float*  topd = (float*) (ws + 32768 + 2 * 8388608);              // 320 KB
  int*    topi = (int*)   (ws + 32768 + 2 * 8388608 + 327680);     // 320 KB
  float*  mask = (float*) (ws + 32768 + 2 * 8388608 + 2 * 327680); // 32 KB
  float*  csc  = (float*) (ws + 32768 + 2 * 8388608 + 2 * 327680 + 32768);

  prep_kernel    <<<NN,            256,  0, stream>>>(x, x2, ahi, alo);
  topk_kernel    <<<NN / 16,       128,  0, stream>>>(ahi, alo, x2, topd, topi);
  zmask_kernel   <<<NN / 256,      256,  0, stream>>>(mask);
  median_kernel  <<<1,             1024, 0, stream>>>(topd, csc);
  scatter_kernel <<<(NN * 10)/256, 256,  0, stream>>>(topi, mask);
  affinity_kernel<<<dim3(NN / 64, NN / 64), 128, 0, stream>>>(ahi, alo, x2, mask, csc, out);
}
